// PointerNetwork_4999341933008
// MI455X (gfx1250) — matrix operand order repaired and checked by reading, not compiled, not run on hardware
//
#include <hip/hip_runtime.h>
#include <hip/hip_bf16.h>

// ---------------------------------------------------------------------------
// PointerNetwork forward for MI455X (gfx1250, wave32, WMMA).
// Two mandatory 256MB streaming passes over h_passage; all GEMMs done with
// v_wmma_f32_16x16x32_bf16 (f32 accumulate), memory-bound by design.
// GEMM inner loop: A stream is register-pipelined one K-step ahead using a
// single raw-register set (pack consumes at issue, loads overwrite after);
// B is issued first each iteration so the pre-WMMA wait leaves the A
// prefetch outstanding (LOADcnt decrements in order). K is a template
// constant so B tile addresses are immediate offsets (no per-iter 64-bit
// pointer arithmetic, minimal VGPR pressure -> no spills).
// ---------------------------------------------------------------------------

typedef __attribute__((ext_vector_type(16))) __bf16    v16bf;
typedef __attribute__((ext_vector_type(8)))  float     v8f;
typedef __attribute__((ext_vector_type(8)))  unsigned  v8u;

__device__ __forceinline__ unsigned short f2bf_bits(float f) {
  unsigned u = __builtin_bit_cast(unsigned, f);
  // round-to-nearest-even to bf16 (used only in one-time weight conversion)
  return (unsigned short)((u + 0x7FFFu + ((u >> 16) & 1u)) >> 16);
}

// Pack two f32 into one packed-bf16 dword (truncation) with a single
// v_perm_b32: dst = {hi[31:16], lo[31:16]}.
__device__ __forceinline__ unsigned pack_bf16(float lo, float hi) {
#if __has_builtin(__builtin_amdgcn_perm)
  return __builtin_amdgcn_perm(__builtin_bit_cast(unsigned, hi),
                               __builtin_bit_cast(unsigned, lo),
                               0x07060302u);
#else
  return (__builtin_bit_cast(unsigned, hi) & 0xFFFF0000u) |
         (__builtin_bit_cast(unsigned, lo) >> 16);
#endif
}

// ---------------------------------------------------------------------------
// Pad/convert a row-major (Arows, K) f32 weight into (NpadRows, K) bf16,
// zero-filling the padding rows so the GEMM inner loop is branch-free.
// ---------------------------------------------------------------------------
__global__ void convert_w_bf16(const float* __restrict__ W,
                               unsigned short* __restrict__ Wb,
                               int Arows, int K, int total) {
  int i = blockIdx.x * blockDim.x + threadIdx.x;
  if (i >= total) return;
  int n = i / K;
  Wb[i] = (n < Arows) ? f2bf_bits(W[i]) : (unsigned short)0;
}

// ---------------------------------------------------------------------------
// Y(M, Npad) = X(M, K) @ Wb(Npad, K)^T
//   X : f32 row-major (streamed, converted to bf16 in registers)
//   Wb: bf16 row-major, rows zero-padded to Npad
// One wave owns a 16-row M tile and NT consecutive 16-col N tiles so the
// streamed A data is loaded once per K-step and reused NT times.
// ---------------------------------------------------------------------------
template <int NT, int K>
__global__ void gemm_bf16_kernel(const float* __restrict__ X,
                                 const unsigned short* __restrict__ Wb,
                                 float* __restrict__ Y,
                                 int M, int Npad) {
  const int lane  = threadIdx.x & 31;
  const int wave  = threadIdx.x >> 5;
  const int mtile = blockIdx.x * (blockDim.x >> 5) + wave;
  if (mtile * 16 >= M) return;               // wave-uniform: EXEC stays full
  const int half = lane >> 4;                // K-half select
  const int l15  = lane & 15;                // row (A) / col (B,C)
  const int n0   = blockIdx.y * (NT * 16);

  // A fragment 16x32 bf16 (contiguous K, matches B):
  //   lanes 0-15 : v0-7 -> K=k0+0..15
  //   lanes16-31 : v0-7 -> K=k0+16..31
  const float* xp = X + (size_t)(mtile * 16 + l15) * K + half * 16;

  // B fragment 32x16: lanes 0-15 hold K=k0..k0+15 of column n,
  // lanes 16-31 hold K=k0+16..k0+31 (row n of Wb is contiguous).
  // Tile t lives at compile-time byte offset t*16*K*2 from bp.
  const unsigned short* bp = Wb + (size_t)(n0 + l15) * K + half * 16;

  const v8f vzero = {0.f, 0.f, 0.f, 0.f, 0.f, 0.f, 0.f, 0.f};
  v8f acc[NT];
#pragma unroll
  for (int t = 0; t < NT; ++t) acc[t] = vzero;

  // ---- preload A for K-step 0 ----
  float4 a0 = *reinterpret_cast<const float4*>(xp);
  float4 a1 = *reinterpret_cast<const float4*>(xp + 4);
  float4 a2 = *reinterpret_cast<const float4*>(xp + 8);
  float4 a3 = *reinterpret_cast<const float4*>(xp + 12);

  const int iters = K / 32;
  for (int it = 0; it < iters; ++it) {
    // (1) B loads for THIS step, issued first: LOADcnt decrements in order,
    //     so the pre-WMMA wait can retire B while the A prefetch (issued
    //     after) stays outstanding for the whole iteration.
    v16bf bfrag[NT];
#pragma unroll
    for (int t = 0; t < NT; ++t)
      bfrag[t] = *reinterpret_cast<const v16bf*>(bp + (size_t)t * 16 * K);

    // (2) pack current A (raw regs are consumed at VALU issue time)
    v8u ap;
    ap[0] = pack_bf16(a0.x, a0.y); ap[1] = pack_bf16(a0.z, a0.w);
    ap[2] = pack_bf16(a1.x, a1.y); ap[3] = pack_bf16(a1.z, a1.w);
    ap[4] = pack_bf16(a2.x, a2.y); ap[5] = pack_bf16(a2.z, a2.w);
    ap[6] = pack_bf16(a3.x, a3.y); ap[7] = pack_bf16(a3.z, a3.w);
    const v16bf afrag = __builtin_bit_cast(v16bf, ap);

    // (3) A prefetch for step it+1 into the SAME raw registers
    if (it + 1 < iters) {
      const float* nx = xp + 32;
      a0 = *reinterpret_cast<const float4*>(nx);
      a1 = *reinterpret_cast<const float4*>(nx + 4);
      a2 = *reinterpret_cast<const float4*>(nx + 8);
      a3 = *reinterpret_cast<const float4*>(nx + 12);
    }

    // (4) WMMA group (waits only for B + last iteration's A)
#pragma unroll
    for (int t = 0; t < NT; ++t) {
      acc[t] = __builtin_amdgcn_wmma_f32_16x16x32_bf16(
          /*neg_a=*/false, afrag, /*neg_b=*/false, bfrag[t],
          /*c_mod=*/(short)0, acc[t], /*reuse_a=*/false, /*reuse_b=*/false);
    }

    xp += 32;
    bp += 32;
  }

  // C layout: acc element r in lane l -> row (8*half + r), col l15.
#pragma unroll
  for (int t = 0; t < NT; ++t) {
    const int n = n0 + t * 16 + l15;
    float* yc = Y + (size_t)(mtile * 16 + half * 8) * Npad + n;
#pragma unroll
    for (int r = 0; r < 8; ++r) yc[(size_t)r * Npad] = acc[t][r];
  }
}

// ---------------------------------------------------------------------------
// s[m] = sum_{n<A} v[n] * tanh(Y[m][n] + (hW ? hW[m%B][n] : 0)) + bias
// transposePB: write out[b*P + p] with m = p*B + b, else out[m].
// ---------------------------------------------------------------------------
__global__ void score_epilogue(const float* __restrict__ Y,
                               const float* __restrict__ hW,
                               const float* __restrict__ v,
                               const float* __restrict__ biasPtr,
                               float* __restrict__ out,
                               int Mrows, int Acols, int Npad,
                               int Bdim, int Pdim, int transposePB) {
  int m = blockIdx.x * blockDim.x + threadIdx.x;
  if (m >= Mrows) return;
  int b = m % Bdim;
  const float* yr = Y + (size_t)m * Npad;
  const float* hr = hW ? hW + (size_t)b * Npad : nullptr;
  float acc = 0.f;
  for (int n = 0; n < Acols; ++n) {
    float t = yr[n] + (hr ? hr[n] : 0.f);
    acc += v[n] * tanhf(t);
  }
  if (biasPtr) acc += biasPtr[0];
  if (transposePB) {
    int p = m / Bdim;
    out[(size_t)b * Pdim + p] = acc;
  } else {
    out[m] = acc;
  }
}

// ---------------------------------------------------------------------------
// Per-batch softmax over Q of s_q[q*B+b], then ha0[b][:] = sum_q a[q]*u[q][b][:]
// One block per batch element; Q == 128, blockDim == 256.
// ---------------------------------------------------------------------------
__global__ void softmax_pool_q(const float* __restrict__ sq,
                               const float* __restrict__ u,
                               float* __restrict__ ha0,
                               int Q, int Bdim, int H) {
  __shared__ float sh[128];
  __shared__ float red[256];
  const int b = blockIdx.x, tid = threadIdx.x;
  float val = -INFINITY;
  if (tid < Q) { sh[tid] = sq[(size_t)tid * Bdim + b]; val = sh[tid]; }
  red[tid] = val;
  __syncthreads();
  for (int s = 128; s > 0; s >>= 1) {
    if (tid < s) red[tid] = fmaxf(red[tid], red[tid + s]);
    __syncthreads();
  }
  const float mx = red[0];
  __syncthreads();
  float e = (tid < Q) ? __expf(sh[tid] - mx) : 0.f;
  red[tid] = e;
  __syncthreads();
  for (int s = 128; s > 0; s >>= 1) {
    if (tid < s) red[tid] += red[tid + s];
    __syncthreads();
  }
  const float Z = red[0];
  __syncthreads();
  if (tid < Q) sh[tid] = e / Z;
  __syncthreads();
  for (int k = tid; k < H; k += blockDim.x) {
    float acc = 0.f;
    for (int q = 0; q < Q; ++q)
      acc += sh[q] * u[((size_t)q * Bdim + b) * H + k];
    ha0[(size_t)b * H + k] = acc;
  }
}

// ---------------------------------------------------------------------------
// Per-batch softmax weights over P. s1 is laid out [b*P + p] (== start logits
// already written to d_out). One block per b, blockDim == 256.
// ---------------------------------------------------------------------------
__global__ void softmax_w1(const float* __restrict__ s1bp,
                           float* __restrict__ w1, int Pdim) {
  __shared__ float red[256];
  const int b = blockIdx.x, tid = threadIdx.x;
  const float* s = s1bp + (size_t)b * Pdim;
  float mx = -INFINITY;
  for (int p = tid; p < Pdim; p += 256) mx = fmaxf(mx, s[p]);
  red[tid] = mx;
  __syncthreads();
  for (int st = 128; st > 0; st >>= 1) {
    if (tid < st) red[tid] = fmaxf(red[tid], red[tid + st]);
    __syncthreads();
  }
  mx = red[0];
  __syncthreads();
  float sum = 0.f;
  for (int p = tid; p < Pdim; p += 256) sum += __expf(s[p] - mx);
  red[tid] = sum;
  __syncthreads();
  for (int st = 128; st > 0; st >>= 1) {
    if (tid < st) red[tid] += red[tid + st];
    __syncthreads();
  }
  const float Z = red[0];
  for (int p = tid; p < Pdim; p += 256)
    w1[(size_t)b * Pdim + p] = __expf(s[p] - mx) / Z;
}

// ---------------------------------------------------------------------------
// Pass 2 over h_passage (256 MB): partial weighted sums (deterministic,
// no atomics). cpart[pc][b][k] = sum over the pc-th P-chunk of w*h.
// grid = (B, PCH), blockDim = 256 (threads cover H=1024 via float4).
// ---------------------------------------------------------------------------
__global__ void weighted_sum_part(const float* __restrict__ h,
                                  const float* __restrict__ w1,
                                  float* __restrict__ cpart,
                                  int Pdim, int Bdim, int H) {
  const int b = blockIdx.x;
  const int pc = blockIdx.y;
  const int chunk = Pdim / gridDim.y;
  const int p0 = pc * chunk;
  const int k4 = threadIdx.x * 4;
  float4 acc = make_float4(0.f, 0.f, 0.f, 0.f);
  for (int p = p0; p < p0 + chunk; ++p) {
    const float w = w1[(size_t)b * Pdim + p];
    const float4 hv =
        *reinterpret_cast<const float4*>(h + ((size_t)p * Bdim + b) * H + k4);
    acc.x += w * hv.x; acc.y += w * hv.y; acc.z += w * hv.z; acc.w += w * hv.w;
  }
  *reinterpret_cast<float4*>(cpart + ((size_t)pc * Bdim + b) * H + k4) = acc;
}

__global__ void reduce_c(const float* __restrict__ cpart,
                         float* __restrict__ c, int nch, int BH) {
  int i = blockIdx.x * blockDim.x + threadIdx.x;
  if (i >= BH) return;
  float s = 0.f;
  for (int pc = 0; pc < nch; ++pc) s += cpart[(size_t)pc * BH + i];
  c[i] = s;
}

// ---------------------------------------------------------------------------
// torch GRUCell (bias=False), gates ordered [r, z, n]; gi/gh are (B, 3H).
// ---------------------------------------------------------------------------
__global__ void gru_combine(const float* __restrict__ gi,
                            const float* __restrict__ gh,
                            const float* __restrict__ h0,
                            float* __restrict__ h1, int Bdim, int H) {
  int idx = blockIdx.x * blockDim.x + threadIdx.x;
  if (idx >= Bdim * H) return;
  int b = idx / H, k = idx % H;
  size_t base = (size_t)b * 3 * H;
  float ir = gi[base + k], iz = gi[base + H + k], in_ = gi[base + 2 * H + k];
  float hr = gh[base + k], hz = gh[base + H + k], hn  = gh[base + 2 * H + k];
  float r = 1.f / (1.f + __expf(-(ir + hr)));
  float z = 1.f / (1.f + __expf(-(iz + hz)));
  float n = tanhf(in_ + r * hn);
  float hprev = h0[(size_t)b * H + k];
  h1[(size_t)b * H + k] = (1.f - z) * n + z * hprev;
}

// ---------------------------------------------------------------------------
extern "C" void kernel_launch(void* const* d_in, const int* in_sizes, int n_in,
                              void* d_out, int out_size, void* d_ws, size_t ws_size,
                              hipStream_t stream) {
  (void)in_sizes; (void)n_in; (void)out_size; (void)ws_size;
  constexpr int Pp = 1024, Qq = 128, Bb = 64, HPc = 1024, HQc = 1024;
  constexpr int Ac = 75, APAD = 80, H3 = 3 * HQc;        // 3072
  constexpr int Mq = Qq * Bb;                            // 8192
  constexpr int Mp = Pp * Bb;                            // 65536
  constexpr int PCH = 16;

  const float* h_passage  = (const float*)d_in[0];
  const float* u_question = (const float*)d_in[1];
  const float* Whp  = (const float*)d_in[2];
  const float* Wha  = (const float*)d_in[3];
  const float* v_w  = (const float*)d_in[4];
  const float* Wuq  = (const float*)d_in[5];
  const float* v1_w = (const float*)d_in[6];
  const float* v1_b = (const float*)d_in[7];
  const float* W_ih = (const float*)d_in[8];
  const float* W_hh = (const float*)d_in[9];
  float* out = (float*)d_out;                 // [start(B*P) | end(B*P)]

  // --- workspace carve (256B aligned) ---
  char* ws = (char*)d_ws;
  size_t off = 0;
  auto carve = [&](size_t bytes) -> void* {
    void* p = ws + off;
    off += (bytes + 255) & ~(size_t)255;
    return p;
  };
  float* tq   = (float*)carve((size_t)Mq * APAD * 4);     // u@Wuq.T
  float* sq   = (float*)carve((size_t)Mq * 4);            // question scores
  float* ha0  = (float*)carve((size_t)Bb * HQc * 4);
  float* hw0  = (float*)carve((size_t)Bb * APAD * 4);     // ha0@Wha.T
  float* Wh   = (float*)carve((size_t)Mp * APAD * 4);     // h@Whp.T (21 MB)
  float* w1   = (float*)carve((size_t)Mp * 4);            // softmax weights
  float* cpart= (float*)carve((size_t)PCH * Bb * HPc * 4);
  float* cvec = (float*)carve((size_t)Bb * HPc * 4);
  float* gi   = (float*)carve((size_t)Bb * H3 * 4);
  float* gh   = (float*)carve((size_t)Bb * H3 * 4);
  float* ha1  = (float*)carve((size_t)Bb * HQc * 4);
  float* hw1  = (float*)carve((size_t)Bb * APAD * 4);
  unsigned short* whp_bf = (unsigned short*)carve((size_t)APAD * HPc * 2);
  unsigned short* wuq_bf = (unsigned short*)carve((size_t)APAD * HQc * 2);
  unsigned short* wha_bf = (unsigned short*)carve((size_t)APAD * HQc * 2);
  unsigned short* wih_bf = (unsigned short*)carve((size_t)H3 * HPc * 2);
  unsigned short* whh_bf = (unsigned short*)carve((size_t)H3 * HQc * 2);

  // --- weights -> bf16 (zero-padded rows) ---
  {
    int tot = APAD * HPc;
    convert_w_bf16<<<(tot + 255) / 256, 256, 0, stream>>>(Whp, whp_bf, Ac, HPc, tot);
    convert_w_bf16<<<(tot + 255) / 256, 256, 0, stream>>>(Wuq, wuq_bf, Ac, HQc, tot);
    convert_w_bf16<<<(tot + 255) / 256, 256, 0, stream>>>(Wha, wha_bf, Ac, HQc, tot);
    int tot3 = H3 * HPc;
    convert_w_bf16<<<(tot3 + 255) / 256, 256, 0, stream>>>(W_ih, wih_bf, H3, HPc, tot3);
    convert_w_bf16<<<(tot3 + 255) / 256, 256, 0, stream>>>(W_hh, whh_bf, H3, HQc, tot3);
  }

  // --- question path: t = u@Wuq.T ; s_q ; softmax-pool -> ha0 ---
  gemm_bf16_kernel<5, 1024><<<dim3(Mq / 16 / 8, 1), 256, 0, stream>>>(
      u_question, wuq_bf, tq, Mq, APAD);
  score_epilogue<<<Mq / 256, 256, 0, stream>>>(
      tq, nullptr, v1_w, v1_b, sq, Mq, Ac, APAD, Bb, Pp, 0);
  softmax_pool_q<<<Bb, 256, 0, stream>>>(sq, u_question, ha0, Qq, Bb, HQc);

  // --- hw0 = ha0@Wha.T ---
  gemm_bf16_kernel<5, 1024><<<dim3(1, 1), 256, 0, stream>>>(ha0, wha_bf, hw0, Bb, APAD);

  // --- pass 1 over h_passage: Wh = h@Whp.T (256 MB stream) ---
  gemm_bf16_kernel<5, 1024><<<dim3(Mp / 16 / 8, 1), 256, 0, stream>>>(
      h_passage, whp_bf, Wh, Mp, APAD);

  // --- s1 -> start logits (transposed to (B,P)) ---
  score_epilogue<<<Mp / 256, 256, 0, stream>>>(
      Wh, hw0, v_w, nullptr, out, Mp, Ac, APAD, Bb, Pp, 1);

  // --- softmax over P, pass 2 over h_passage -> c ---
  softmax_w1<<<Bb, 256, 0, stream>>>(out, w1, Pp);
  weighted_sum_part<<<dim3(Bb, PCH), 256, 0, stream>>>(h_passage, w1, cpart, Pp, Bb, HPc);
  reduce_c<<<(Bb * HPc + 255) / 256, 256, 0, stream>>>(cpart, cvec, PCH, Bb * HPc);

  // --- GRU: gi = c@W_ih.T, gh = ha0@W_hh.T, combine -> ha1 ---
  gemm_bf16_kernel<2, 1024><<<dim3(1, H3 / 32), 256, 0, stream>>>(cvec, wih_bf, gi, Bb, H3);
  gemm_bf16_kernel<2, 1024><<<dim3(1, H3 / 32), 256, 0, stream>>>(ha0, whh_bf, gh, Bb, H3);
  gru_combine<<<(Bb * HQc + 255) / 256, 256, 0, stream>>>(gi, gh, ha0, ha1, Bb, HQc);

  // --- hw1 = ha1@Wha.T ; s2 -> end logits ---
  gemm_bf16_kernel<5, 1024><<<dim3(1, 1), 256, 0, stream>>>(ha1, wha_bf, hw1, Bb, APAD);
  score_epilogue<<<Mp / 256, 256, 0, stream>>>(
      Wh, hw1, v_w, nullptr, out + (size_t)Bb * Pp, Mp, Ac, APAD, Bb, Pp, 1);
}
